// AdaptiveStateMixer_52295521796514
// MI455X (gfx1250) — matrix operand order repaired and checked by reading, not compiled, not run on hardware
//
#include <hip/hip_runtime.h>
#include <hip/hip_bf16.h>

typedef __attribute__((ext_vector_type(16))) __bf16 v16bf;
typedef __attribute__((ext_vector_type(8)))  __bf16 v8bf;
typedef __attribute__((ext_vector_type(2)))  __bf16 v2bf;
typedef __attribute__((ext_vector_type(8)))  float  v8f;
typedef __attribute__((ext_vector_type(4)))  float  v4f;

#define BDIM 8
#define SDIM 2048
#define HDIM 1024

union Frag { v16bf v; v8bf h[2]; v2bf p2[8]; };
union Pack8 { v8bf v; v2bf p[4]; };

// Hardware packed f32->bf16 conversion (V_CVT_PK_BF16_F32 on gfx1250).
__device__ __forceinline__ v2bf f2bf2(float a, float b) {
#if __has_builtin(__builtin_amdgcn_cvt_pk_bf16_f32)
  auto p = __builtin_amdgcn_cvt_pk_bf16_f32(a, b);
  union { decltype(p) x; v2bf r; } u;
  u.x = p;
  return u.r;
#else
  v2bf r;
  r[0] = (__bf16)a;
  r[1] = (__bf16)b;
  return r;
#endif
}
__device__ __forceinline__ __bf16 f2bf(float f) { return f2bf2(f, f)[0]; }

// A fragment (16x32 bf16): lane<16 holds K{0..15}, lane>=16 holds K{16..31}
__device__ __forceinline__ v16bf load_frag_a(const __bf16* rowbase, int half) {
  Frag f;
  f.h[0] = *(const v8bf*)(rowbase + 16 * half);
  f.h[1] = *(const v8bf*)(rowbase + 16 * half + 8);
  return f.v;
}
// B fragment (32x16 bf16): lane = column, K rows 16*half..16*half+15 contiguous
__device__ __forceinline__ v16bf load_frag_b(const __bf16* colbase, int half) {
  Frag f;
  f.h[0] = *(const v8bf*)(colbase + 16 * half);
  f.h[1] = *(const v8bf*)(colbase + 16 * half + 8);
  return f.v;
}
__device__ __forceinline__ v8f wmma_bf16(v16bf a, v16bf b, v8f c) {
  return __builtin_amdgcn_wmma_f32_16x16x32_bf16(false, a, false, b, (short)0, c,
                                                 false, false);
}
__device__ __forceinline__ float gelu_exact(float x) {
  return 0.5f * x * (1.0f + erff(x * 0.70710678118654752f));
}

// ---------------------------------------------------------------------------
// Generic projection GEMM: C = act(A[M,K] * W[K,N] + bias)
// MODE 0: bf16 row-major out (no act)     -> outb[row*N+col]
// MODE 1: gelu, bf16 transposed per batch -> outb[(b*N+col)*S + s]   (Vt)
// MODE 2: gelu, fp32 out                  -> outf[row*N+col]
// A_F32: A is fp32 (convert while staging) else bf16.
// Compile-time specialization keeps the K-loop straight-line (no per-iter
// SALU branching) so vmem/ds clauses and wmma overlap stay tight.
// ---------------------------------------------------------------------------
template <int MODE, bool A_F32>
__global__ __launch_bounds__(256) void proj_gemm(
    const float* __restrict__ A32, const __bf16* __restrict__ A16,
    const float* __restrict__ W, const float* __restrict__ bias,
    int M, int N, int K,
    __bf16* __restrict__ outb, float* __restrict__ outf) {
  __shared__ __bf16 As[128][40];   // 128x32 tile, pitch 40 (80B, 16B aligned)
  __shared__ __bf16 Bs[64][40];    // transposed: Bs[n][k]

  int tid = threadIdx.x;
  int wv = tid >> 5, lane = tid & 31;
  int half = lane >> 4, r = lane & 15;
  int mw = wv & 3, nw = wv >> 2;            // 4 M-groups x 2 N-groups of 32x32
  int m0 = blockIdx.x * 128;
  int n0 = blockIdx.y * 64;

  v8f z = {};
  v8f acc[2][2] = {{z, z}, {z, z}};

  int arow = tid >> 1;
  int acol = (tid & 1) * 16;
  int brow = tid >> 3;
  int bcol = (tid & 7) * 8;

  for (int kk = 0; kk < K; kk += 32) {
    // ---- stage A tile (128x32) as bf16
    if (A_F32) {
      const float* src = A32 + (size_t)(m0 + arow) * K + kk + acol;
      #pragma unroll
      for (int c = 0; c < 2; ++c) {
        v4f lo = *(const v4f*)(src + c * 8);
        v4f hi = *(const v4f*)(src + c * 8 + 4);
        Pack8 tmp;
        tmp.p[0] = f2bf2(lo[0], lo[1]);
        tmp.p[1] = f2bf2(lo[2], lo[3]);
        tmp.p[2] = f2bf2(hi[0], hi[1]);
        tmp.p[3] = f2bf2(hi[2], hi[3]);
        *(v8bf*)&As[arow][acol + c * 8] = tmp.v;
      }
    } else {
      const __bf16* src = A16 + (size_t)(m0 + arow) * K + kk + acol;
      *(v8bf*)&As[arow][acol]     = *(const v8bf*)src;
      *(v8bf*)&As[arow][acol + 8] = *(const v8bf*)(src + 8);
    }
    // ---- stage B tile (32x64) transposed -> Bs[n][k]
    {
      const float* src = W + (size_t)(kk + brow) * N + n0 + bcol;
      #pragma unroll
      for (int j = 0; j < 8; ++j) Bs[bcol + j][brow] = f2bf(src[j]);
    }
    __syncthreads();

    v16bf bfr[2];
    #pragma unroll
    for (int tj = 0; tj < 2; ++tj)
      bfr[tj] = load_frag_b(&Bs[nw * 32 + tj * 16 + r][0], half);
    #pragma unroll
    for (int ti = 0; ti < 2; ++ti) {
      v16bf a = load_frag_a(&As[mw * 32 + ti * 16 + r][0], half);
      acc[ti][0] = wmma_bf16(a, bfr[0], acc[ti][0]);
      acc[ti][1] = wmma_bf16(a, bfr[1], acc[ti][1]);
    }
    __syncthreads();
  }

  // ---- epilogue (C/D layout: VGPR i -> M = 8*half+i, N = r)
  #pragma unroll
  for (int ti = 0; ti < 2; ++ti)
  #pragma unroll
  for (int tj = 0; tj < 2; ++tj) {
    int col = n0 + nw * 32 + tj * 16 + r;
    float bi = bias[col];
    #pragma unroll
    for (int i = 0; i < 8; ++i) {
      int row = m0 + mw * 32 + ti * 16 + 8 * half + i;
      float x = acc[ti][tj][i] + bi;
      if (MODE == 0) {
        outb[(size_t)row * N + col] = f2bf(x);
      } else if (MODE == 1) {
        x = gelu_exact(x);
        int bidx = row >> 11;          // / S
        int srow = row & (SDIM - 1);   // % S
        outb[((size_t)bidx * N + col) * SDIM + srow] = f2bf(x);
      } else {
        outf[(size_t)row * N + col] = gelu_exact(x);
      }
    }
  }
}

// ---------------------------------------------------------------------------
// Attention: one workgroup = one batch b, 32 queries [q0, q0+32).
// Full 32x2048 fp32 score row-block in LDS (267KB; CDNA5 WGP has 320KB).
// ---------------------------------------------------------------------------
__global__ __launch_bounds__(256) void attn_kernel(
    const __bf16* __restrict__ Qb, const __bf16* __restrict__ Kb,
    const __bf16* __restrict__ Vt, __bf16* __restrict__ Rb,
    const float* __restrict__ time_seq, const unsigned char* __restrict__ pad,
    const float* __restrict__ order_scale, const float* __restrict__ time_scale,
    float* __restrict__ attn_out) {
  extern __shared__ float smem[];
  float* sc     = smem;             // 32*2048 fp32 scores
  float* tks    = sc + 32 * SDIM;   // 2048 time values
  float* red    = tks + SDIM;       // 32x8 partial reductions
  float* rowm   = red + 256;        // 32 row max
  float* rowsum = rowm + 32;        // 32 row sum
  unsigned char* pads = (unsigned char*)(rowsum + 32);  // 2048 pad flags

  int tid = threadIdx.x;
  int wv = tid >> 5, lane = tid & 31;
  int half = lane >> 4, r = lane & 15;
  int mh = wv & 1, nq = wv >> 1;     // 2 q-halves x 4 key/h quarters
  int b  = blockIdx.x >> 6;          // S/32 = 64 query blocks per batch
  int q0 = (blockIdx.x & 63) * 32;

  #pragma unroll
  for (int j = 0; j < 8; ++j) {
    int idx = tid * 8 + j;
    tks[idx]  = time_seq[(size_t)b * SDIM + idx];
    pads[idx] = pad[(size_t)b * SDIM + idx];
  }
  float sp_o = log1pf(expf(order_scale[0]));
  float sp_t = log1pf(expf(time_scale[0]));
  __syncthreads();

  // ---------------- score phase: S = Q K^T / sqrt(H) + biases, masked -------
  const __bf16* Qrow = Qb + ((size_t)b * SDIM + q0 + 16 * mh + r) * HDIM;
  for (int kt = 0; kt < SDIM; kt += 256) {
    int kbase = kt + nq * 64;
    v8f z = {};
    v8f acc[4] = {z, z, z, z};
    for (int h0 = 0; h0 < HDIM; h0 += 32) {
      v16bf a = load_frag_a(Qrow + h0, half);
      #pragma unroll
      for (int nt = 0; nt < 4; ++nt) {
        const __bf16* kp =
            Kb + ((size_t)b * SDIM + kbase + nt * 16 + r) * HDIM + h0;
        acc[nt] = wmma_bf16(a, load_frag_b(kp, half), acc[nt]);
      }
    }
    #pragma unroll
    for (int nt = 0; nt < 4; ++nt) {
      int key = kbase + nt * 16 + r;
      #pragma unroll
      for (int i = 0; i < 8; ++i) {
        int ql = 16 * mh + 8 * half + i;
        int qg = q0 + ql;
        float s;
        if (key >= qg || pads[key]) {        // strictly causal + padding
          s = -1.0e30f;
        } else {
          float raw = acc[nt][i] * 0.03125f; // 1/sqrt(1024)
          float d   = (float)(qg - key);
          float tg  = fmaxf(tks[qg] - tks[key], 0.0f);
          s = raw - sp_o * __logf(1.0f + d) - sp_t * __logf(1.0f + tg);
        }
        sc[ql * SDIM + key] = s;
      }
    }
  }
  __syncthreads();

  // ---------------- softmax stats: 8 threads per row, vectorized ------------
  int rrow = tid >> 3, seg = tid & 7;
  {
    float mx = -3.4e38f;
    const v4f* p = (const v4f*)(sc + rrow * SDIM + seg * 256);
    for (int j = 0; j < 64; ++j) {
      v4f t = p[j];
      mx = fmaxf(mx, fmaxf(fmaxf(t[0], t[1]), fmaxf(t[2], t[3])));
    }
    red[rrow * 8 + seg] = mx;
  }
  __syncthreads();
  if (seg == 0) {
    float mx = red[rrow * 8];
    for (int j = 1; j < 8; ++j) mx = fmaxf(mx, red[rrow * 8 + j]);
    rowm[rrow] = mx;
  }
  __syncthreads();
  {
    float m = rowm[rrow], sm = 0.0f;
    const v4f* p = (const v4f*)(sc + rrow * SDIM + seg * 256);
    for (int j = 0; j < 64; ++j) {
      v4f t = p[j];
      sm += __expf(t[0] - m) + __expf(t[1] - m) +
            __expf(t[2] - m) + __expf(t[3] - m);
    }
    red[rrow * 8 + seg] = sm;
  }
  __syncthreads();
  if (seg == 0) {
    float sm = 0.0f;
    for (int j = 0; j < 8; ++j) sm += red[rrow * 8 + j];
    rowsum[rrow] = sm;
  }
  __syncthreads();

  // ---------------- write normalized attn (fp32, 16B per lane, coalesced) ---
  float* aout = attn_out + ((size_t)b * SDIM + q0) * SDIM;
  for (int iter = 0; iter < 64; ++iter) {
    int idx = iter * 1024 + tid * 4;           // 4 consecutive keys, same row
    int ql = idx >> 11, key = idx & (SDIM - 1);
    float m = rowm[ql];
    float scale = (m <= -0.5e30f) ? 0.0f : (1.0f / rowsum[ql]);
    v4f t = *(const v4f*)(sc + idx);
    v4f o;
    o[0] = __expf(t[0] - m) * scale;
    o[1] = __expf(t[1] - m) * scale;
    o[2] = __expf(t[2] - m) * scale;
    o[3] = __expf(t[3] - m) * scale;
    *(v4f*)(aout + (size_t)ql * SDIM + key) = o;
  }

  // ---------------- attn @ V : P(bf16, built on the fly) x Vt ---------------
  {
    int ql = 16 * mh + r;
    float m = rowm[ql];
    float scale = (m <= -0.5e30f) ? 0.0f : (1.0f / rowsum[ql]);
    const float* srow = sc + ql * SDIM;
    v8f z = {};
    v8f acc2[16];
    #pragma unroll
    for (int t = 0; t < 16; ++t) acc2[t] = z;

    for (int k0 = 0; k0 < SDIM; k0 += 32) {
      Frag af;
      const float* c0 = srow + k0 + 16 * half;
      const float* c1 = srow + k0 + 16 * half + 8;
      #pragma unroll
      for (int j = 0; j < 4; ++j)
        af.p2[j] = f2bf2(__expf(c0[2 * j] - m) * scale,
                         __expf(c0[2 * j + 1] - m) * scale);
      #pragma unroll
      for (int j = 0; j < 4; ++j)
        af.p2[4 + j] = f2bf2(__expf(c1[2 * j] - m) * scale,
                             __expf(c1[2 * j + 1] - m) * scale);
      #pragma unroll
      for (int nt = 0; nt < 16; ++nt) {
        const __bf16* vp =
            Vt + ((size_t)b * HDIM + nq * 256 + nt * 16 + r) * SDIM +
            k0 + 16 * half;
        Frag bf_;
        bf_.h[0] = *(const v8bf*)vp;
        bf_.h[1] = *(const v8bf*)(vp + 8);
        acc2[nt] = wmma_bf16(af.v, bf_.v, acc2[nt]);
      }
    }
    #pragma unroll
    for (int nt = 0; nt < 16; ++nt) {
      int hcol = nq * 256 + nt * 16 + r;
      #pragma unroll
      for (int i = 0; i < 8; ++i) {
        int qrow = q0 + 16 * mh + 8 * half + i;
        Rb[((size_t)b * SDIM + qrow) * HDIM + hcol] = f2bf(acc2[nt][i]);
      }
    }
  }
}

// ---------------------------------------------------------------------------
extern "C" void kernel_launch(void* const* d_in, const int* in_sizes, int n_in,
                              void* d_out, int out_size, void* d_ws,
                              size_t ws_size, hipStream_t stream) {
  (void)in_sizes; (void)n_in; (void)out_size; (void)ws_size;
  const float* query       = (const float*)d_in[0];
  const float* memory      = (const float*)d_in[1];
  const float* time_seq    = (const float*)d_in[2];
  const float* Wq          = (const float*)d_in[3];
  const float* bq          = (const float*)d_in[4];
  const float* Wk          = (const float*)d_in[5];
  const float* bk          = (const float*)d_in[6];
  const float* Wv          = (const float*)d_in[7];
  const float* bv          = (const float*)d_in[8];
  const float* Wo          = (const float*)d_in[9];
  const float* bo          = (const float*)d_in[10];
  const float* order_scale = (const float*)d_in[11];
  const float* time_scale  = (const float*)d_in[12];
  const unsigned char* pad = (const unsigned char*)d_in[13];

  const int M = BDIM * SDIM;           // 16384
  size_t MB = (size_t)M * HDIM;        // 16,777,216 elems
  __bf16* Qb = (__bf16*)d_ws;          // 32MB each, 128MB total ws
  __bf16* Kb = Qb + MB;
  __bf16* Vt = Kb + MB;                // V transposed: [b][h][s]
  __bf16* Rb = Vt + MB;                // retrieved (pre out-proj), bf16
  float* out = (float*)d_out;          // retrieved fp32 [B,S,H]
  float* attn_out = out + MB;          // attn fp32 [B,S,S]

  dim3 ggrid(M / 128, HDIM / 64);
  proj_gemm<0, true><<<ggrid, 256, 0, stream>>>(query,  nullptr, Wq, bq,
                                                M, HDIM, HDIM, Qb, nullptr);
  proj_gemm<0, true><<<ggrid, 256, 0, stream>>>(memory, nullptr, Wk, bk,
                                                M, HDIM, HDIM, Kb, nullptr);
  proj_gemm<1, true><<<ggrid, 256, 0, stream>>>(memory, nullptr, Wv, bv,
                                                M, HDIM, HDIM, Vt, nullptr);

  size_t shbytes = (size_t)(32 * SDIM + SDIM + 256 + 32 + 32) * sizeof(float) + SDIM;
  (void)hipFuncSetAttribute(reinterpret_cast<const void*>(attn_kernel),
                            hipFuncAttributeMaxDynamicSharedMemorySize,
                            (int)shbytes);
  attn_kernel<<<dim3(BDIM * (SDIM / 32)), 256, shbytes, stream>>>(
      Qb, Kb, Vt, Rb, time_seq, pad, order_scale, time_scale, attn_out);

  proj_gemm<2, false><<<ggrid, 256, 0, stream>>>(nullptr, Rb, Wo, bo,
                                                 M, HDIM, HDIM, nullptr, out);
}
